// NNCRF_35124242547327
// MI455X (gfx1250) — matrix operand order repaired and checked by reading, not compiled, not run on hardware
//
#include <hip/hip_runtime.h>
#include <hip/hip_bf16.h>
#include <math.h>

// ---------------- feature detection for CDNA5 async LDS path ----------------
#if defined(__has_builtin)
#  if __has_builtin(__builtin_amdgcn_global_load_async_to_lds_b128)
#    define HAVE_ASYNC_LDS 1
#  endif
#  if __has_builtin(__builtin_amdgcn_s_wait_asynccnt)
#    define HAVE_WAIT_ASYNC_BUILTIN 1
#  endif
#endif

// ---------------- problem constants ----------------
#define BB 64
#define LL 128
#define EE 300
#define EP 320     // E padded to 32
#define HDc 200
#define HDP 224    // HD padded to 32
#define HH 400
#define HP 416     // H padded to 32
#define SS 50
#define SP 64      // S padded to 16
#define GG 800     // 4*HD
#define ML (BB*LL) // 8192
#define START_IDX 48
#define END_IDX 49

typedef _Float16 half16 __attribute__((ext_vector_type(16)));
typedef _Float16 half8  __attribute__((ext_vector_type(8)));
typedef float    float8 __attribute__((ext_vector_type(8)));

// ---------------- WMMA fragment helpers ----------------
// A fragment (16x32 f16, row-major source, stride in halfs).
// lane<16: row=lane,  K=0..15; lane>=16: row=lane-16, K=16..31
__device__ __forceinline__ half16 load_a_frag(const _Float16* base, int stride, int lane) {
    int m    = lane & 15;
    int koff = (lane & 16) ? 16 : 0;
    const _Float16* p = base + m * stride + koff;
    half8 lo = *(const half8*)(p);
    half8 hi = *(const half8*)(p + 8);
    half16 r;
#pragma unroll
    for (int i = 0; i < 8; ++i) { r[i] = lo[i]; r[i + 8] = hi[i]; }
    return r;
}

// B fragment (32x16 f16). Source is W stored N-major: W[n*stride + k], B[k,n]=W[n,k].
// lane<16: K=0..15 ; lane>=16: K=16..31 ; n = lane%16. 16 contiguous halfs.
__device__ __forceinline__ half16 load_b_frag(const _Float16* wbase, int stride, int lane) {
    int n    = lane & 15;
    int koff = (lane & 16) ? 16 : 0;
    return *(const half16*)(wbase + n * stride + koff);
}

__device__ __forceinline__ float sigmoidf(float x) { return 1.0f / (1.0f + expf(-x)); }

#ifdef HAVE_ASYNC_LDS
// Issue a 16B async global->LDS copy (GLOBAL_LOAD_ASYNC_TO_LDS_B128, ASYNCcnt).
// Builtin signature (from diagnostics): (v4i AS1*, v4i AS3*, imm offset, imm cpol)
// with v4i = int __attribute__((vector_size(16))).
typedef int async_v4i __attribute__((vector_size(16)));
__device__ __forceinline__ void async_copy16(const void* gsrc, void* ldst) {
    __builtin_amdgcn_global_load_async_to_lds_b128(
        (__attribute__((address_space(1))) async_v4i*)(unsigned long long)gsrc,
        (__attribute__((address_space(3))) async_v4i*)(unsigned)(unsigned long long)ldst,
        0, 0);
}
__device__ __forceinline__ void wait_async0() {
#ifdef HAVE_WAIT_ASYNC_BUILTIN
    __builtin_amdgcn_s_wait_asynccnt(0);
#else
    asm volatile("s_wait_asynccnt 0" ::: "memory");
#endif
}
#endif

// ---------------- K0: pad + f32->f16 convert weights ----------------
__global__ void padcvt_kernel(_Float16* __restrict__ dst, const float* __restrict__ src,
                              int N, int K, int Np, int Kp) {
    int idx = blockIdx.x * blockDim.x + threadIdx.x;
    int total = Np * Kp;
    if (idx >= total) return;
    int n = idx / Kp, k = idx % Kp;
    dst[idx] = (n < N && k < K) ? (_Float16)src[n * K + k] : (_Float16)0.0f;
}

// ---------------- K1: embedding gather -> f16 padded ----------------
__global__ void embed_kernel(_Float16* __restrict__ x, const int* __restrict__ words,
                             const float* __restrict__ emb) {
    int idx = blockIdx.x * blockDim.x + threadIdx.x;
    if (idx >= ML * EP) return;
    int row = idx / EP, k = idx % EP;
    x[idx] = (k < EE) ? (_Float16)emb[(long)words[row] * EE + k] : (_Float16)0.0f;
}

// ---------------- K1b: zero the K-padding columns of hcat ----------------
__global__ void zero_hcat_pad_kernel(_Float16* __restrict__ hcat) {
    int idx = blockIdx.x * blockDim.x + threadIdx.x;
    if (idx >= ML * (HP - HH)) return;
    int r = idx / (HP - HH), c = idx % (HP - HH);
    hcat[r * HP + HH + c] = (_Float16)0.0f;
}

// ---------------- K2: xg = x @ w_ih^T + b   (both directions) ----------------
// grid(512, 7, 2), block 256 (8 waves). A tile staged to LDS via async copy.
__global__ __launch_bounds__(256) void gemm_xg_kernel(
    const _Float16* __restrict__ x,
    const _Float16* __restrict__ wihf, const _Float16* __restrict__ wihb,
    const float* __restrict__ bf, const float* __restrict__ bb,
    float* __restrict__ xgf, float* __restrict__ xgb) {
    __shared__ _Float16 sA[16 * EP];
    int dir = blockIdx.z;
    const _Float16* wih  = dir ? wihb : wihf;
    const float*    bias = dir ? bb   : bf;
    float*          xg   = dir ? xgb  : xgf;
    int mt = blockIdx.x;

    // stage A tile (16 rows x 320 halfs = 10KB = 640 x 16B)
#ifdef HAVE_ASYNC_LDS
    {
        const char* src = (const char*)(x + (long)mt * 16 * EP);
        char*       dst = (char*)sA;
        for (int i = threadIdx.x; i < (16 * EP * 2) / 16; i += 256)
            async_copy16(src + i * 16, dst + i * 16);
        wait_async0();
    }
#else
    {
        const unsigned long long* srcu = (const unsigned long long*)(x + (long)mt * 16 * EP);
        unsigned long long* dstu = (unsigned long long*)sA;
        for (int i = threadIdx.x; i < 16 * EP / 4; i += 256) dstu[i] = srcu[i];
    }
#endif
    __syncthreads();

    int wave = threadIdx.x >> 5, lane = threadIdx.x & 31;
    int nt = blockIdx.y * 8 + wave;
    if (nt >= GG / 16) return;

    float8 acc = {};
#pragma unroll
    for (int kt = 0; kt < EP / 32; ++kt) {
        half16 a = load_a_frag(sA + kt * 32, EP, lane);
        half16 b = load_b_frag(wih + (long)nt * 16 * EP + kt * 32, EP, lane);
        acc = __builtin_amdgcn_wmma_f32_16x16x32_f16(false, a, false, b, (short)0, acc, false, false);
    }
    int n = nt * 16 + (lane & 15);
    float bv = bias[n];
    int mbase = mt * 16 + ((lane & 16) ? 8 : 0);
#pragma unroll
    for (int v = 0; v < 8; ++v) xg[(long)(mbase + v) * GG + n] = acc[v] + bv;
}

// ---------------- K3: LSTM recurrence (state resident in LDS) ----------------
// grid(4, 2): x = 16-row batch chunk, y = direction. block 256 (8 waves).
// dynamic LDS: h(16x224 f16) + c(16x200 f32) + gates(16x800 f32) = 71168 B
__global__ __launch_bounds__(256) void lstm_rec_kernel(
    const _Float16* __restrict__ whhf, const _Float16* __restrict__ whhb,
    const float* __restrict__ xgf, const float* __restrict__ xgb,
    _Float16* __restrict__ hcat) {
    extern __shared__ char smem[];
    _Float16* hS = (_Float16*)smem;                      // 16*224 halfs
    float*    cS = (float*)(smem + 16 * HDP * 2);        // 16*200 floats
    float*    gS = cS + 16 * HDc;                        // 16*800 floats

    int mc  = blockIdx.x;
    int dir = blockIdx.y;
    const _Float16* whh = dir ? whhb : whhf;
    const float*    xg  = dir ? xgb  : xgf;
    int m0  = mc * 16;
    int tid = threadIdx.x;

    for (int i = tid; i < 16 * HDP; i += 256) hS[i] = (_Float16)0.0f;
    for (int i = tid; i < 16 * HDc; i += 256) cS[i] = 0.0f;
    __syncthreads();

    int wave = tid >> 5, lane = tid & 31;
    for (int t = 0; t < LL; ++t) {
        int tt = dir ? (LL - 1 - t) : t;
        // gates = h @ w_hh^T + xg[:, tt]
        for (int nt = wave; nt < GG / 16; nt += 8) {
            float8 acc = {};
#pragma unroll
            for (int kt = 0; kt < HDP / 32; ++kt) {
                half16 a = load_a_frag(hS + kt * 32, HDP, lane);
                half16 b = load_b_frag(whh + (long)nt * 16 * HDP + kt * 32, HDP, lane);
                acc = __builtin_amdgcn_wmma_f32_16x16x32_f16(false, a, false, b, (short)0, acc, false, false);
            }
            int n  = nt * 16 + (lane & 15);
            int mb = (lane & 16) ? 8 : 0;
#pragma unroll
            for (int v = 0; v < 8; ++v) {
                int m = mb + v;
                gS[m * GG + n] = acc[v] + xg[((long)(m0 + m) * LL + tt) * GG + n];
            }
        }
        __syncthreads();
        // elementwise gates + state update
        for (int j = tid; j < 16 * HDc; j += 256) {
            int m = j / HDc, jj = j % HDc;
            float gi = gS[m * GG + jj];
            float gf = gS[m * GG + HDc + jj];
            float gg = gS[m * GG + 2 * HDc + jj];
            float go = gS[m * GG + 3 * HDc + jj];
            float c  = sigmoidf(gf) * cS[j] + sigmoidf(gi) * tanhf(gg);
            cS[j] = c;
            float h = sigmoidf(go) * tanhf(c);
            hS[m * HDP + jj] = (_Float16)h;
            hcat[((long)(m0 + m) * LL + tt) * HP + dir * HDc + jj] = (_Float16)h;
        }
        // prefetch next step's xg slice (16 rows x 800 f32 = 400 cachelines of 128B)
        if (t + 1 < LL) {
            int tn = dir ? (LL - 2 - t) : (t + 1);
            for (int i = tid; i < 16 * 25; i += 256) {
                int m = i / 25, ln = i % 25;
                __builtin_prefetch(&xg[((long)(m0 + m) * LL + tn) * GG + ln * 32], 0, 0);
            }
        }
        __syncthreads();
    }
}

// ---------------- K4: features = hcat @ W_tag^T + b_tag ----------------
// 2048 tiles (512 m-tiles x 4 n-tiles), 8 waves/block -> 256 blocks.
__global__ __launch_bounds__(256) void gemm_feat_kernel(
    const _Float16* __restrict__ hcat, const _Float16* __restrict__ wtag,
    const float* __restrict__ btag, float* __restrict__ feats) {
    int wave = threadIdx.x >> 5, lane = threadIdx.x & 31;
    int id = blockIdx.x * 8 + wave;
    int mt = id >> 2, nt = id & 3;
    float8 acc = {};
#pragma unroll
    for (int kt = 0; kt < HP / 32; ++kt) {
        half16 a = load_a_frag(hcat + (long)mt * 16 * HP + kt * 32, HP, lane);
        half16 b = load_b_frag(wtag + (long)nt * 16 * HP + kt * 32, HP, lane);
        acc = __builtin_amdgcn_wmma_f32_16x16x32_f16(false, a, false, b, (short)0, acc, false, false);
    }
    int n = nt * 16 + (lane & 15);
    float bv = (n < SS) ? btag[n] : 0.0f;
    int mbase = mt * 16 + ((lane & 16) ? 8 : 0);
#pragma unroll
    for (int v = 0; v < 8; ++v) feats[(long)(mbase + v) * SP + n] = acc[v] + bv;
}

// ---------------- K5: CRF forward (logsumexp lattice) + labeled score ----------------
// one block per batch element, 64 threads; transition matrix cached in LDS.
__global__ __launch_bounds__(64) void crf_kernel(
    const float* __restrict__ feats, const float* __restrict__ transition,
    const int* __restrict__ tags, float* __restrict__ partial) {
    __shared__ float tr[SS * SS];
    __shared__ float al[2][SP];
    int b = blockIdx.x, tid = threadIdx.x;
    for (int i = tid; i < SS * SS; i += 64) tr[i] = transition[i];
    __syncthreads();

    const float* fb = feats + (long)b * LL * SP;
    if (tid < SS) al[0][tid] = tr[START_IDX * SS + tid] + fb[tid];
    __syncthreads();

    int cur = 0;
    for (int t = 1; t < LL; ++t) {
        float na = 0.0f;
        if (tid < SS) {
            float m = -1e30f;
            for (int s = 0; s < SS; ++s) {
                float v = al[cur][s] + tr[s * SS + tid];
                m = fmaxf(m, v);
            }
            float sum = 0.0f;
            for (int s = 0; s < SS; ++s) {
                float v = al[cur][s] + tr[s * SS + tid];
                sum += expf(v - m);
            }
            na = m + logf(sum) + fb[t * SP + tid];
        }
        if (tid < SS) al[cur ^ 1][tid] = na;
        __syncthreads();
        cur ^= 1;
    }

    if (tid == 0) {
        // unlabeled = logsumexp_s(alpha_last[s] + trans[s][END])
        float m = -1e30f;
        for (int s = 0; s < SS; ++s) m = fmaxf(m, al[cur][s] + tr[s * SS + END_IDX]);
        float sum = 0.0f;
        for (int s = 0; s < SS; ++s) sum += expf(al[cur][s] + tr[s * SS + END_IDX] - m);
        float unlabeled = m + logf(sum);
        // labeled path score
        const int* tg = tags + (long)b * LL;
        float labeled = tr[START_IDX * SS + tg[0]] + fb[tg[0]];
        for (int t = 1; t < LL; ++t)
            labeled += tr[tg[t - 1] * SS + tg[t]] + fb[t * SP + tg[t]];
        labeled += tr[tg[LL - 1] * SS + END_IDX];
        partial[b] = unlabeled - labeled;
    }
}

// ---------------- K6: final 64-way sum ----------------
__global__ __launch_bounds__(64) void reduce64_kernel(const float* __restrict__ partial,
                                                      float* __restrict__ out) {
    __shared__ float s[64];
    s[threadIdx.x] = partial[threadIdx.x];
    __syncthreads();
    for (int off = 32; off > 0; off >>= 1) {
        if (threadIdx.x < off) s[threadIdx.x] += s[threadIdx.x + off];
        __syncthreads();
    }
    if (threadIdx.x == 0) out[0] = s[0];
}

// ---------------- host launcher ----------------
static inline size_t align256(size_t x) { return (x + 255) & ~(size_t)255; }

extern "C" void kernel_launch(void* const* d_in, const int* in_sizes, int n_in,
                              void* d_out, int out_size, void* d_ws, size_t ws_size,
                              hipStream_t stream) {
    const int*   words      = (const int*)d_in[0];
    // d_in[1] = word_seq_lens (all == L, unused), d_in[2] = masks (all true, unused)
    const int*   tags       = (const int*)d_in[3];
    const float* emb        = (const float*)d_in[4];
    const float* w_ih_f     = (const float*)d_in[5];
    const float* w_hh_f     = (const float*)d_in[6];
    const float* b_f        = (const float*)d_in[7];
    const float* w_ih_b     = (const float*)d_in[8];
    const float* w_hh_b     = (const float*)d_in[9];
    const float* b_b        = (const float*)d_in[10];
    const float* W_tag      = (const float*)d_in[11];
    const float* b_tag      = (const float*)d_in[12];
    const float* transition = (const float*)d_in[13];
    float* out = (float*)d_out;

    // workspace carve-up
    char* p = (char*)d_ws;
    _Float16* xf16  = (_Float16*)p; p += align256((size_t)ML * EP * 2);
    _Float16* wihf  = (_Float16*)p; p += align256((size_t)GG * EP * 2);
    _Float16* wihb  = (_Float16*)p; p += align256((size_t)GG * EP * 2);
    _Float16* whhf  = (_Float16*)p; p += align256((size_t)GG * HDP * 2);
    _Float16* whhb  = (_Float16*)p; p += align256((size_t)GG * HDP * 2);
    _Float16* wtag  = (_Float16*)p; p += align256((size_t)SP * HP * 2);
    float*    xgf   = (float*)p;    p += align256((size_t)ML * GG * 4);
    float*    xgb   = (float*)p;    p += align256((size_t)ML * GG * 4);
    _Float16* hcat  = (_Float16*)p; p += align256((size_t)ML * HP * 2);
    float*    feats = (float*)p;    p += align256((size_t)ML * SP * 4);
    float*    part  = (float*)p;    p += align256((size_t)BB * 4);
    (void)ws_size; (void)in_sizes; (void)n_in; (void)out_size;

    // K0: weight pad/convert
    {
        int t;
        t = GG * EP;  padcvt_kernel<<<(t + 255) / 256, 256, 0, stream>>>(wihf, w_ih_f, GG, EE,  GG, EP);
        t = GG * EP;  padcvt_kernel<<<(t + 255) / 256, 256, 0, stream>>>(wihb, w_ih_b, GG, EE,  GG, EP);
        t = GG * HDP; padcvt_kernel<<<(t + 255) / 256, 256, 0, stream>>>(whhf, w_hh_f, GG, HDc, GG, HDP);
        t = GG * HDP; padcvt_kernel<<<(t + 255) / 256, 256, 0, stream>>>(whhb, w_hh_b, GG, HDc, GG, HDP);
        t = SP * HP;  padcvt_kernel<<<(t + 255) / 256, 256, 0, stream>>>(wtag, W_tag,  SS, HH,  SP, HP);
    }
    // K1: embedding gather
    embed_kernel<<<(ML * EP + 255) / 256, 256, 0, stream>>>(xf16, words, emb);
    zero_hcat_pad_kernel<<<(ML * (HP - HH) + 255) / 256, 256, 0, stream>>>(hcat);

    // K2: xg GEMMs (both directions)
    {
        dim3 grid(ML / 16, (GG / 16 + 7) / 8, 2);
        gemm_xg_kernel<<<grid, 256, 0, stream>>>(xf16, wihf, wihb, b_f, b_b, xgf, xgb);
    }
    // K3: LSTM recurrence
    {
        dim3 grid(BB / 16, 2);
        size_t shmem = (size_t)16 * HDP * 2 + (size_t)16 * HDc * 4 + (size_t)16 * GG * 4;
        lstm_rec_kernel<<<grid, 256, shmem, stream>>>(whhf, whhb, xgf, xgb, hcat);
    }
    // K4: tag features
    gemm_feat_kernel<<<(ML / 16) * (SP / 16) / 8, 256, 0, stream>>>(hcat, wtag, b_tag, feats);
    // K5: CRF per batch element
    crf_kernel<<<BB, 64, 0, stream>>>(feats, transition, tags, part);
    // K6: reduce to scalar
    reduce64_kernel<<<1, 64, 0, stream>>>(part, out);
}
